// Model_23441931501711
// MI455X (gfx1250) — matrix operand order repaired and checked by reading, not compiled, not run on hardware
//
#include <hip/hip_runtime.h>
#include <hip/hip_bf16.h>
#include <math.h>

// ---------------------------------------------------------------------------
// Shapes: B=8, T=12, N=1024, D=64, S=24, SAMPLES=30.  BT = 96, ROWS = 98304.
// ---------------------------------------------------------------------------
#define BT      96
#define NNODE   1024
#define DDIM    64
#define NEIGH   24
#define SAMP    30
#define SPAD    32
#define ROWS    (BT * NNODE)          // 98304
#define NE      ((size_t)ROWS * DDIM) // 6291456
#define NTILES  (ROWS / 16)           // 6144

typedef __attribute__((ext_vector_type(16))) _Float16 v16h;
typedef __attribute__((ext_vector_type(8)))  _Float16 v8h;
typedef __attribute__((ext_vector_type(8)))  float    v8f;
typedef __attribute__((ext_vector_type(4)))  float    v4f;
typedef __attribute__((ext_vector_type(2)))  float    v2f;

// ---------------------------------------------------------------------------
// Wave32 reductions
// ---------------------------------------------------------------------------
__device__ inline float wred_sum(float v) {
#pragma unroll
  for (int off = 16; off > 0; off >>= 1) v += __shfl_xor(v, off, 32);
  return v;
}
__device__ inline float wred_max(float v) {
#pragma unroll
  for (int off = 16; off > 0; off >>= 1) v = fmaxf(v, __shfl_xor(v, off, 32));
  return v;
}

// ---------------------------------------------------------------------------
// WMMA helper (gfx1250: V_WMMA_F32_16X16X32_F16)
// ---------------------------------------------------------------------------
__device__ inline v8f wmma_f16(v16h a, v16h b, v8f c) {
  return __builtin_amdgcn_wmma_f32_16x16x32_f16(false, a, false, b, (short)0, c,
                                                false, false);
}

// A fragment (16x32 f16) from f32 row; lane row = lane&15, khalf = lane>>4.
// Elements 0..15 <- K [kb+16*khalf, +16) (contiguous).
// Both runs 32B-aligned -> global_load_b128 x4.
__device__ inline v16h load_a_frag(const float* __restrict__ row, int kb, int khalf) {
  const v4f* p0 = (const v4f*)(row + kb + khalf * 16);
  const v4f* p1 = (const v4f*)(row + kb + 8 + khalf * 16);
  v4f r0 = p0[0], r1 = p0[1], r2 = p1[0], r3 = p1[1];
  v16h a;
#pragma unroll
  for (int j = 0; j < 4; ++j) {
    a[j]      = (_Float16)r0[j];
    a[4 + j]  = (_Float16)r1[j];
    a[8 + j]  = (_Float16)r2[j];
    a[12 + j] = (_Float16)r3[j];
  }
  return a;
}

// B fragment (32x16 f16), lane col = lane&15, element j = K row kb+16*khalf+j,
// from a (K x ld) row-major f32 weight (stride-ld gather).
__device__ inline v16h load_b_frag_w(const float* __restrict__ W, int col, int kb,
                                     int khalf, int ld) {
  v16h b;
  int k0 = kb + khalf * 16;
#pragma unroll
  for (int j = 0; j < 16; ++j) b[j] = (_Float16)W[(size_t)(k0 + j) * ld + col];
  return b;
}

// B fragment where B[k][col] = krow_col[k] (multiplying by K^T; contiguous k).
__device__ inline v16h load_b_frag_rowk(const float* __restrict__ krow, int kb,
                                        int khalf) {
  const v4f* p = (const v4f*)(krow + kb + khalf * 16);
  v4f r0 = p[0], r1 = p[1], r2 = p[2], r3 = p[3];
  v16h b;
#pragma unroll
  for (int j = 0; j < 4; ++j) {
    b[j]      = (_Float16)r0[j];
    b[4 + j]  = (_Float16)r1[j];
    b[8 + j]  = (_Float16)r2[j];
    b[12 + j] = (_Float16)r3[j];
  }
  return b;
}

// ---------------------------------------------------------------------------
// 1) x_ = x + eigvec[d,n,0] * eigvalue[d]
// ---------------------------------------------------------------------------
__global__ void xprep_kernel(const float* __restrict__ x,
                             const float* __restrict__ eigvec,
                             const float* __restrict__ eigval,
                             float* __restrict__ xo) {
  size_t i = (size_t)blockIdx.x * blockDim.x + threadIdx.x;
  if (i >= NE) return;
  int d = (int)(i & 63);
  int n = (int)((i >> 6) & 1023);
  xo[i] = x[i] + eigvec[d * NNODE + n] * eigval[d];
}

// ---------------------------------------------------------------------------
// 2) 98304x64 @ 64x64 WMMA GEMM: C = act(A @ W + bias [+ resid]).
//    Template flags compiled out of the epilogue.  STORE_TR additionally
//    emits a d-major f16 transposed copy (Vt[bt][d][k]) via one packed b128
//    store per lane (lane owns a column, 8 consecutive rows).
//    grid-stride over 16-row tiles; block = 128 (4 waves, wave = ntile);
//    weight fragments built once per wave, reused across all tiles.
// ---------------------------------------------------------------------------
template <int RELU, int HAS_RES, int STORE_TR>
__global__ void gemm64_kernel(const float* __restrict__ A,
                              const float* __restrict__ W,
                              const float* __restrict__ bias,
                              const float* __restrict__ resid,
                              float* __restrict__ Cout,
                              _Float16* __restrict__ vt) {
  int ntile = threadIdx.x >> 5;
  int lane  = threadIdx.x & 31;
  int khalf = lane >> 4;
  int lcol  = lane & 15;
  int col   = ntile * 16 + lcol;

  v16h b0 = load_b_frag_w(W, col, 0, khalf, DDIM);
  v16h b1 = load_b_frag_w(W, col, 32, khalf, DDIM);
  float bcol = bias[col];

  for (int tile = blockIdx.x; tile < NTILES; tile += gridDim.x) {
    const float* arow = A + (size_t)(tile * 16 + lcol) * DDIM;
    v16h a0 = load_a_frag(arow, 0, khalf);
    v16h a1 = load_a_frag(arow, 32, khalf);

    v8f c = {};
    c = wmma_f16(a0, b0, c);
    c = wmma_f16(a1, b1, c);

    float vals[8];
#pragma unroll
    for (int v = 0; v < 8; ++v) {
      int row = tile * 16 + khalf * 8 + v;
      float val = c[v] + bcol;
      if (HAS_RES) val += resid[(size_t)row * DDIM + col];
      if (RELU) val = fmaxf(val, 0.f);
      vals[v] = val;
      Cout[(size_t)row * DDIM + col] = val;
    }
    if (STORE_TR) {
      v8h h;
#pragma unroll
      for (int v = 0; v < 8; ++v) h[v] = (_Float16)vals[v];
      int btv   = (tile * 16) >> 10;                 // 16 | 1024 -> no split
      int krow0 = ((tile * 16) & 1023) + khalf * 8;  // 8 consecutive k
      *(v8h*)(vt + ((size_t)btv * DDIM + col) * NNODE + krow0) = h;
    }
  }
}

// ---------------------------------------------------------------------------
// 3) Neighbor gather attention -> pooled score M.  One wave per node.
// ---------------------------------------------------------------------------
__global__ void gat_kernel(const float* __restrict__ Q, const float* __restrict__ K,
                           const float* __restrict__ V, const int* __restrict__ adj,
                           const float* __restrict__ Wp, const float* __restrict__ bp,
                           float* __restrict__ M) {
  int gwave = (int)((blockIdx.x * blockDim.x + threadIdx.x) >> 5);
  int lane  = threadIdx.x & 31;
  if (gwave >= ROWS) return;
  int bt = gwave >> 10;
  int n  = gwave & 1023;
  v2f q = ((const v2f*)(Q + (size_t)gwave * DDIM))[lane];
  float g0 = 0.f, g1 = 0.f;
  size_t base = (size_t)bt * NNODE;
  for (int s = 0; s < NEIGH; ++s) {
    int a = adj[n * NEIGH + s];
    v2f kv = ((const v2f*)(K + (base + a) * DDIM))[lane];
    v2f vv = ((const v2f*)(V + (base + a) * DDIM))[lane];
    float p = wred_sum(q[0] * kv[0] + q[1] * kv[1]);
    g0 += p * vv[0];
    g1 += p * vv[1];
  }
  v2f wp = ((const v2f*)Wp)[lane];
  float m = wred_sum(g0 * wp[0] + g1 * wp[1]);
  if (lane == 0) M[gwave] = m + bp[0];
}

// ---------------------------------------------------------------------------
// 4) Top-30 per (b,t): 30 rounds of block argmax (set semantics suffice:
//    the cp-gather result is invariant to ordering of the selected set).
// ---------------------------------------------------------------------------
__global__ void topk_kernel(const float* __restrict__ M, int* __restrict__ topidx) {
  __shared__ float vals[NNODE];
  __shared__ float rmax[256];
  __shared__ int   ridx[256];
  int bt = blockIdx.x;
  int t  = threadIdx.x;
  for (int j = t; j < NNODE; j += 256) vals[j] = M[(size_t)bt * NNODE + j];
  __syncthreads();
  for (int it = 0; it < SAMP; ++it) {
    float bestv = -INFINITY;
    int   besti = 0;
    for (int j = t; j < NNODE; j += 256) {
      float v = vals[j];
      if (v > bestv) { bestv = v; besti = j; }
    }
    rmax[t] = bestv; ridx[t] = besti;
    __syncthreads();
    for (int stride = 128; stride > 0; stride >>= 1) {
      if (t < stride) {
        float v2 = rmax[t + stride];
        int   i2 = ridx[t + stride];
        if (v2 > rmax[t] || (v2 == rmax[t] && i2 < ridx[t])) {
          rmax[t] = v2; ridx[t] = i2;
        }
      }
      __syncthreads();
    }
    if (t == 0) { topidx[bt * SPAD + it] = ridx[0]; vals[ridx[0]] = -INFINITY; }
    __syncthreads();
  }
  if (t < 2) topidx[bt * SPAD + SAMP + t] = 0;
}

// ---------------------------------------------------------------------------
// 5) scores[bt, s, n] = dot(Q[topidx[s]], K[n]) / 8  via WMMA.
//    grid = BT*64, block = 64 (2 waves, wave = mtile); rows s>=30 -> zero A.
// ---------------------------------------------------------------------------
__global__ void scores_kernel(const float* __restrict__ Q, const float* __restrict__ K,
                              const int* __restrict__ topidx,
                              float* __restrict__ scores) {
  int blk   = blockIdx.x;
  int bt    = blk >> 6;
  int ntile = blk & 63;
  int mtile = threadIdx.x >> 5;
  int lane  = threadIdx.x & 31;
  int khalf = lane >> 4, lcol = lane & 15;

  int s = mtile * 16 + lcol;
  v16h a0 = {}, a1 = {};
  if (s < SAMP) {
    int node = topidx[bt * SPAD + s];
    const float* qrow = Q + ((size_t)bt * NNODE + node) * DDIM;
    a0 = load_a_frag(qrow, 0, khalf);
    a1 = load_a_frag(qrow, 32, khalf);
  }
  int n = ntile * 16 + lcol;
  const float* krow = K + ((size_t)bt * NNODE + n) * DDIM;
  v16h b0 = load_b_frag_rowk(krow, 0, khalf);
  v16h b1 = load_b_frag_rowk(krow, 32, khalf);

  v8f c = {};
  c = wmma_f16(a0, b0, c);
  c = wmma_f16(a1, b1, c);

#pragma unroll
  for (int v = 0; v < 8; ++v) {
    int row = mtile * 16 + khalf * 8 + v;
    scores[((size_t)bt * SPAD + row) * NNODE + ntile * 16 + lcol] = c[v] * 0.125f;
  }
}

// ---------------------------------------------------------------------------
// 6) Row softmax over n (s<30), zero rows 30/31, cp[n] = argmax_s attn[s,n].
// ---------------------------------------------------------------------------
__global__ void softmax_cp_kernel(float* __restrict__ attn, int* __restrict__ cp) {
  int bt   = blockIdx.x;
  int wave = threadIdx.x >> 5;
  int lane = threadIdx.x & 31;
  float* base = attn + (size_t)bt * SPAD * NNODE;
  for (int s = wave; s < SAMP; s += 8) {
    float* row = base + (size_t)s * NNODE;
    float m = -INFINITY;
    for (int j = lane; j < NNODE; j += 32) m = fmaxf(m, row[j]);
    m = wred_max(m);
    float z = 0.f;
    for (int j = lane; j < NNODE; j += 32) {
      float e = __expf(row[j] - m);
      row[j] = e;
      z += e;
    }
    z = wred_sum(z);
    float inv = 1.f / z;
    for (int j = lane; j < NNODE; j += 32) row[j] *= inv;
  }
  for (int j = threadIdx.x; j < 2 * NNODE; j += 256) base[SAMP * NNODE + j] = 0.f;
  __syncthreads();
  for (int n = threadIdx.x; n < NNODE; n += 256) {
    float bv = base[n];
    int   bi = 0;
    for (int s = 1; s < SAMP; ++s) {
      float v = base[(size_t)s * NNODE + n];
      if (v > bv) { bv = v; bi = s; }
    }
    cp[(size_t)bt * NNODE + n] = bi;
  }
}

// ---------------------------------------------------------------------------
// 7) av = attn(32x1024) @ V(1024x64) via WMMA.  B operand comes from the
//    pre-transposed f16 Vt[bt][d][k]: one contiguous 32B v16h load per step.
//    One block per (b,t), 8 waves: wave -> (mtile = w>>2, ntile = w&3).
// ---------------------------------------------------------------------------
__global__ void av_kernel(const float* __restrict__ attn,
                          const _Float16* __restrict__ Vt,
                          float* __restrict__ av) {
  int bt    = blockIdx.x;
  int wave  = threadIdx.x >> 5;
  int lane  = threadIdx.x & 31;
  int mtile = wave >> 2, ntile = wave & 3;
  int khalf = lane >> 4, lcol = lane & 15;

  const float* arow = attn + ((size_t)bt * SPAD + mtile * 16 + lcol) * NNODE;
  int d = ntile * 16 + lcol;
  const _Float16* vtrow = Vt + ((size_t)bt * DDIM + d) * NNODE;

  v8f c = {};
  for (int kb = 0; kb < NNODE; kb += 32) {
    v16h a = load_a_frag(arow, kb, khalf);
    v16h b = *(const v16h*)(vtrow + kb + khalf * 16);
    c = wmma_f16(a, b, c);
  }
#pragma unroll
  for (int v = 0; v < 8; ++v) {
    int row = mtile * 16 + khalf * 8 + v;
    av[((size_t)bt * SPAD + row) * DDIM + d] = c[v];
  }
}

// ---------------------------------------------------------------------------
// 8) avWo = av @ Wo + bo  (tiny per-(b,t) WMMA GEMM; gather happens after).
// ---------------------------------------------------------------------------
__global__ void avwo_kernel(const float* __restrict__ av, const float* __restrict__ Wo,
                            const float* __restrict__ bo, float* __restrict__ avwo) {
  int bt    = blockIdx.x;
  int wave  = threadIdx.x >> 5;
  int lane  = threadIdx.x & 31;
  int mtile = wave >> 2, ntile = wave & 3;
  int khalf = lane >> 4, lcol = lane & 15;

  const float* arow = av + ((size_t)bt * SPAD + mtile * 16 + lcol) * DDIM;
  int col = ntile * 16 + lcol;
  v16h b0 = load_b_frag_w(Wo, col, 0, khalf, DDIM);
  v16h b1 = load_b_frag_w(Wo, col, 32, khalf, DDIM);
  v16h a0 = load_a_frag(arow, 0, khalf);
  v16h a1 = load_a_frag(arow, 32, khalf);
  v8f c = {};
  c = wmma_f16(a0, b0, c);
  c = wmma_f16(a1, b1, c);
#pragma unroll
  for (int v = 0; v < 8; ++v) {
    int row = mtile * 16 + khalf * 8 + v;
    avwo[((size_t)bt * SPAD + row) * DDIM + col] = c[v] + bo[col];
  }
}

// ---------------------------------------------------------------------------
// 9) out = layernorm(avWo[cp[n]] + x_[n]).  One wave per row (2 elems/lane).
// ---------------------------------------------------------------------------
__global__ void gather_res_ln_kernel(const float* __restrict__ avwo,
                                     const int* __restrict__ cp,
                                     const float* __restrict__ xr_,
                                     float* __restrict__ out) {
  int gwave = (int)((blockIdx.x * blockDim.x + threadIdx.x) >> 5);
  int lane  = threadIdx.x & 31;
  if (gwave >= ROWS) return;
  int bt  = gwave >> 10;
  int sel = cp[gwave];
  v2f sv = ((const v2f*)(avwo + ((size_t)bt * SPAD + sel) * DDIM))[lane];
  v2f xv = ((const v2f*)(xr_ + (size_t)gwave * DDIM))[lane];
  float v0 = sv[0] + xv[0], v1 = sv[1] + xv[1];
  float mean = wred_sum(v0 + v1) * (1.f / 64.f);
  float d0 = v0 - mean, d1 = v1 - mean;
  float var = wred_sum(d0 * d0 + d1 * d1) * (1.f / 64.f);
  float inv = rsqrtf(var + 1e-5f);
  v2f o; o[0] = d0 * inv; o[1] = d1 * inv;
  ((v2f*)(out + (size_t)gwave * DDIM))[lane] = o;
}

// ---------------------------------------------------------------------------
// 10) Final layernorm (residual already folded by the FFN2 epilogue).
// ---------------------------------------------------------------------------
__global__ void ln_kernel(const float* __restrict__ in, float* __restrict__ out) {
  int gwave = (int)((blockIdx.x * blockDim.x + threadIdx.x) >> 5);
  int lane  = threadIdx.x & 31;
  if (gwave >= ROWS) return;
  v2f r = ((const v2f*)(in + (size_t)gwave * DDIM))[lane];
  float mean = wred_sum(r[0] + r[1]) * (1.f / 64.f);
  float d0 = r[0] - mean, d1 = r[1] - mean;
  float var = wred_sum(d0 * d0 + d1 * d1) * (1.f / 64.f);
  float inv = rsqrtf(var + 1e-5f);
  v2f o; o[0] = d0 * inv; o[1] = d1 * inv;
  ((v2f*)(out + (size_t)gwave * DDIM))[lane] = o;
}

// ---------------------------------------------------------------------------
// Launch
// ---------------------------------------------------------------------------
extern "C" void kernel_launch(void* const* d_in, const int* in_sizes, int n_in,
                              void* d_out, int out_size, void* d_ws, size_t ws_size,
                              hipStream_t stream) {
  const float* x      = (const float*)d_in[0];
  const int*   adj    = (const int*)d_in[1];
  const float* eigvec = (const float*)d_in[2];
  const float* eigval = (const float*)d_in[3];
  const float* Wq = (const float*)d_in[4];  const float* bq = (const float*)d_in[5];
  const float* Wk = (const float*)d_in[6];  const float* bk = (const float*)d_in[7];
  const float* Wv = (const float*)d_in[8];  const float* bv = (const float*)d_in[9];
  const float* Wo = (const float*)d_in[10]; const float* bo = (const float*)d_in[11];
  const float* Wp = (const float*)d_in[12]; const float* bp = (const float*)d_in[13];
  const float* W1 = (const float*)d_in[14]; const float* b1 = (const float*)d_in[15];
  const float* W2 = (const float*)d_in[16]; const float* b2 = (const float*)d_in[17];

  float* ws = (float*)d_ws;
  float* Xp = ws;                                   // NE
  float* Qb = Xp + NE;                              // NE
  float* Kb = Qb + NE;                              // NE
  float* Vb = Kb + NE;                              // NE
  float* Ob = Vb + NE;                              // NE
  float* SC = Ob + NE;                              // BT*32*1024
  float* AV = SC + (size_t)BT * SPAD * NNODE;       // BT*32*64
  float* AW = AV + (size_t)BT * SPAD * DDIM;        // BT*32*64
  float* Mb = AW + (size_t)BT * SPAD * DDIM;        // ROWS
  int*   CP = (int*)(Mb + ROWS);                    // ROWS ints
  int*   TI = CP + ROWS;                            // BT*32 ints
  _Float16* Vt = (_Float16*)(TI + BT * SPAD);       // NE halfs (d-major V)
  float* T1 = Qb;   // FFN hidden reuses Q buffer
  float* HP = Kb;   // FFN pre-LN output reuses K buffer
  (void)in_sizes; (void)n_in; (void)out_size; (void)ws_size;

  const int gblk = 768;  // 8 tiles per block

  // 1) x_ = x + ev * eigvalue
  xprep_kernel<<<(int)((NE + 255) / 256), 256, 0, stream>>>(x, eigvec, eigval, Xp);

  // 2) Q, K, V projections (WMMA); V also emits transposed f16 copy Vt
  gemm64_kernel<0,0,0><<<gblk, 128, 0, stream>>>(Xp, Wq, bq, nullptr, Qb, nullptr);
  gemm64_kernel<0,0,0><<<gblk, 128, 0, stream>>>(Xp, Wk, bk, nullptr, Kb, nullptr);
  gemm64_kernel<0,0,1><<<gblk, 128, 0, stream>>>(Xp, Wv, bv, nullptr, Vb, Vt);

  // 3) Neighbor gather attention -> M scores
  gat_kernel<<<ROWS / 8, 256, 0, stream>>>(Qb, Kb, Vb, adj, Wp, bp, Mb);

  // 4) Top-30 indices per (b,t)
  topk_kernel<<<BT, 256, 0, stream>>>(Mb, TI);

  // 5) scores = Qr @ K^T / 8 (WMMA)
  scores_kernel<<<BT * 64, 64, 0, stream>>>(Qb, Kb, TI, SC);

  // 6) softmax over n, cp = argmax over s
  softmax_cp_kernel<<<BT, 256, 0, stream>>>(SC, CP);

  // 7) av = attn @ V (WMMA, f16 transposed V)
  av_kernel<<<BT, 256, 0, stream>>>(SC, Vt, AV);

  // 8) avWo = av @ Wo + bo (WMMA)
  avwo_kernel<<<BT, 256, 0, stream>>>(AV, Wo, bo, AW);

  // 9) out = LN(avWo[cp] + x_)
  gather_res_ln_kernel<<<ROWS / 8, 256, 0, stream>>>(AW, CP, Xp, Ob);

  // 10) FFN: t1 = relu(out@W1+b1); h = t1@W2+b2+out; final LN
  gemm64_kernel<1,0,0><<<gblk, 128, 0, stream>>>(Ob, W1, b1, nullptr, T1, nullptr);
  gemm64_kernel<0,1,0><<<gblk, 128, 0, stream>>>(T1, W2, b2, Ob, HP, nullptr);
  ln_kernel<<<ROWS / 8, 256, 0, stream>>>(HP, (float*)d_out);
}
